// XFADS_18056042512500
// MI455X (gfx1250) — compile-verified, matrix operand order checked by reading, not run on hardware
//
#include <hip/hip_runtime.h>
#include <hip/hip_bf16.h>
#include <math.h>

// ---------------------------------------------------------------------------
// XFADS on MI455X (gfx1250, wave32, WMMA 16x16x32 f16 -> f32)
//   0) weight convert/transpose to f16 [N][K]
//   1) encoder: fused 2-GEMM WMMA, outputs a-features in f16
//   2) backward RNN: persistent WG, LDS-resident weights, async-to-LDS
//      double-buffered a_t staging, ping-pong h state
//   3) forward scan: persistent WG, async-to-LDS double-buffered alpha,
//      WMMA dynamics MLP + rcp-based natural-parameter update
// ---------------------------------------------------------------------------

typedef _Float16 v16h __attribute__((ext_vector_type(16)));
typedef _Float16 v8h  __attribute__((ext_vector_type(8)));
typedef float    v8f  __attribute__((ext_vector_type(8)));
typedef float    v4f  __attribute__((ext_vector_type(4)));
typedef int      v4i  __attribute__((ext_vector_type(4)));

#define NB   32
#define TT   1024
#define DOBS 128
#define DZ   64
#define DU   8
#define DA   128
#define HENC 256
#define HBW  128
#define HDYN 256
#define KDYN 96   // (DZ+DU)=72 padded to multiple of 32

#if __has_builtin(__builtin_amdgcn_global_load_async_to_lds_b128)
#define HAVE_ASYNC_LDS 1
#else
#define HAVE_ASYNC_LDS 0
#endif

#if HAVE_ASYNC_LDS
#if __has_builtin(__builtin_amdgcn_s_wait_asynccnt)
#define WAIT_ASYNC(N) __builtin_amdgcn_s_wait_asynccnt(N)
#else
#define WAIT_ASYNC(N) asm volatile("s_wait_asynccnt %0" ::"i"(N) : "memory")
#endif
#else
#define WAIT_ASYNC(N) do {} while (0)
#endif

// 16 bytes global -> LDS (async when available, sync fallback)
__device__ __forceinline__ void copy16_g2l(void* lds_dst, const void* g_src) {
#if HAVE_ASYNC_LDS
  __builtin_amdgcn_global_load_async_to_lds_b128(
      (v4i*)(uintptr_t)g_src, (v4i*)lds_dst, 0, 0);
#else
  *(v4f*)lds_dst = *(const v4f*)g_src;
#endif
}

__device__ __forceinline__ v8f wmma32(v16h a, v16h b, v8f c) {
  return __builtin_amdgcn_wmma_f32_16x16x32_f16(
      false, a, false, b, (short)0, c, false, false);
}

// Fragment loader for 16x16x32 f16 WMMA.
//  A operand: row-major [M][K]   (lane&15 = M)
//  B operand: transposed [N][K]  (lane&15 = N)
// halves 0..7 = K{kb..kb+7}, halves 8..15 = K{16+kb..16+kb+7}, kb=8*(lane>>4)
__device__ __forceinline__ v16h load_frag(const _Float16* __restrict__ src, int ld) {
  const int lane = threadIdx.x & 31;
  const int i  = lane & 15;
  const int kb = (lane >> 4) << 3;
  const _Float16* p = src + i * ld + kb;
  v8h lo = *(const v8h*)(p);
  v8h hi = *(const v8h*)(p + 16);
  v16h r;
#pragma unroll
  for (int e = 0; e < 8; ++e) { r[e] = lo[e]; r[8 + e] = hi[e]; }
  return r;
}

__device__ __forceinline__ float fast_tanh(float x) {
#if __has_builtin(__builtin_amdgcn_tanhf)
  return __builtin_amdgcn_tanhf(x);          // V_TANH_F32 (gfx1250 TRANS op)
#elif __has_builtin(__builtin_amdgcn_tanh_f32)
  return __builtin_amdgcn_tanh_f32(x);
#else
  float cx = fminf(fmaxf(x, -15.f), 15.f);   // branch-free fallback
  float e  = __expf(2.0f * cx);
  return 1.0f - 2.0f * __builtin_amdgcn_rcpf(e + 1.0f);
#endif
}

__device__ __forceinline__ float softplus_f(float x) {
  float sp = __logf(1.f + __expf(fminf(x, 20.f)));
  return (x > 20.f) ? x : sp;                // v_cndmask, no divergence
}

// --------------------------------------------------------------------------
// Kernel 0: f32 [K][N] -> f16 [N][Kp] transpose/convert (zero-pad K -> Kp)
// --------------------------------------------------------------------------
__global__ void conv_transpose_f16(const float* __restrict__ src,
                                   _Float16* __restrict__ dst,
                                   int K, int Nn, int Kp) {
  int idx = blockIdx.x * blockDim.x + threadIdx.x;
  if (idx >= Nn * Kp) return;
  int n = idx / Kp, k = idx - n * Kp;
  dst[idx] = (k < K) ? (_Float16)src[k * Nn + n] : (_Float16)0.f;
}

// --------------------------------------------------------------------------
// Kernel 1: encoder  a = tanh(y@W1+b1)@W2+b2 ; rows = N*T ; a stored f16
// --------------------------------------------------------------------------
__global__ __launch_bounds__(512, 1)
void encoder_kernel(const float* __restrict__ y,
                    const _Float16* __restrict__ w1T,   // [256][128]
                    const float* __restrict__ b1,
                    const _Float16* __restrict__ w2T,   // [128][256]
                    const float* __restrict__ b2,
                    _Float16* __restrict__ outA) {      // (N*T,128) f16
  __shared__ _Float16 sY[64 * DOBS];
  __shared__ _Float16 sH[64 * HENC];
  const int tid = threadIdx.x, wave = tid >> 5, lane = tid & 31;
  const int row0 = blockIdx.x * 64;

  for (int i = tid; i < 64 * DOBS; i += 512) {
    int r = i >> 7, d = i & 127;
    sY[i] = (_Float16)y[(size_t)(row0 + r) * DOBS + d];
  }
  __syncthreads();

  // GEMM1: 64x256, 64 tiles, 4 per wave, K=128
#pragma unroll
  for (int tt = 0; tt < 4; ++tt) {
    int ti = wave * 4 + tt;
    int rt = ti >> 4, ct = ti & 15;
    v8f c = {};
#pragma unroll
    for (int kk = 0; kk < 4; ++kk)
      c = wmma32(load_frag(sY + rt * 16 * DOBS + kk * 32, DOBS),
                 load_frag(w1T + ct * 16 * DOBS + kk * 32, DOBS), c);
    int col = ct * 16 + (lane & 15);
    int r0  = rt * 16 + ((lane >> 4) << 3);
    float bb = b1[col];
#pragma unroll
    for (int r = 0; r < 8; ++r)
      sH[(r0 + r) * HENC + col] = (_Float16)fast_tanh(c[r] + bb);
  }
  __syncthreads();

  // GEMM2: 64x128, 32 tiles, 2 per wave, K=256
#pragma unroll
  for (int tt = 0; tt < 2; ++tt) {
    int ti = wave * 2 + tt;
    int rt = ti >> 3, ct = ti & 7;
    v8f c = {};
#pragma unroll
    for (int kk = 0; kk < 8; ++kk)
      c = wmma32(load_frag(sH + rt * 16 * HENC + kk * 32, HENC),
                 load_frag(w2T + ct * 16 * HENC + kk * 32, HENC), c);
    int col = ct * 16 + (lane & 15);
    int r0  = rt * 16 + ((lane >> 4) << 3);
    float bb = b2[col];
#pragma unroll
    for (int r = 0; r < 8; ++r)
      outA[(size_t)(row0 + r0 + r) * DA + col] = (_Float16)(c[r] + bb);
  }
}

// --------------------------------------------------------------------------
// Kernel 2: backward RNN, single persistent WG, async double-buffered a_t.
// --------------------------------------------------------------------------
__global__ __launch_bounds__(512, 1)
void backward_kernel(const _Float16* __restrict__ A16,   // (N*T,128) f16
                     const _Float16* __restrict__ waT,
                     const _Float16* __restrict__ whT,
                     const float* __restrict__ bb,
                     const _Float16* __restrict__ woT,
                     const float* __restrict__ bo,
                     float* __restrict__ alpha) {        // (T,N,128) f32
  extern __shared__ _Float16 smem[];
  _Float16* sWa = smem;
  _Float16* sWh = sWa + 16384;
  _Float16* sWo = sWh + 16384;
  _Float16* sA[2] = {sWo + 16384, sWo + 16384 + 4096};
  _Float16* sH0 = sA[1] + 4096;
  _Float16* sH1 = sH0 + 4096;
  const int tid = threadIdx.x, wave = tid >> 5, lane = tid & 31;

  // stage one (N=32 x 128) f16 a-slice: thread tid owns 8 contiguous halves
  auto stage_a = [&](_Float16* dst, int t) {
    const int n = tid >> 4, d = (tid & 15) * 8;
    copy16_g2l(dst + tid * 8, A16 + (size_t)n * TT * DA + (size_t)t * DA + d);
  };

  for (int i = tid; i < 16384; i += 512) {
    sWa[i] = waT[i]; sWh[i] = whT[i]; sWo[i] = woT[i];
  }
  for (int i = tid; i < 4096; i += 512) sH0[i] = (_Float16)0.f;

  stage_a(sA[0], TT - 1);
  int p = 0;
  _Float16* hPrev = sH0;
  _Float16* hCur  = sH1;

  for (int t = TT - 1; t >= 0; --t) {
    if (t > 0) { stage_a(sA[1 - p], t - 1); WAIT_ASYNC(1); }
    else       { WAIT_ASYNC(0); }
    __syncthreads();

    { // hCur = tanh(a@Wa + hPrev@Wh + b): 16 tiles, 1 per wave
      int rt = wave >> 3, ct = wave & 7;
      v8f c = {};
#pragma unroll
      for (int kk = 0; kk < 4; ++kk)
        c = wmma32(load_frag(sA[p] + rt * 16 * DA + kk * 32, DA),
                   load_frag(sWa + ct * 16 * DA + kk * 32, DA), c);
#pragma unroll
      for (int kk = 0; kk < 4; ++kk)
        c = wmma32(load_frag(hPrev + rt * 16 * HBW + kk * 32, HBW),
                   load_frag(sWh + ct * 16 * HBW + kk * 32, HBW), c);
      int col = ct * 16 + (lane & 15);
      int r0  = rt * 16 + ((lane >> 4) << 3);
      float b0 = bb[col];
#pragma unroll
      for (int r = 0; r < 8; ++r)
        hCur[(r0 + r) * HBW + col] = (_Float16)fast_tanh(c[r] + b0);
    }
    __syncthreads();

    { // b_t = hCur@Wo + bo, constrain, store f32
      int rt = wave >> 3, ct = wave & 7;
      v8f c = {};
#pragma unroll
      for (int kk = 0; kk < 4; ++kk)
        c = wmma32(load_frag(hCur + rt * 16 * HBW + kk * 32, HBW),
                   load_frag(sWo + ct * 16 * HBW + kk * 32, HBW), c);
      int col = ct * 16 + (lane & 15);
      int r0  = rt * 16 + ((lane >> 4) << 3);
      float b0 = bo[col];
#pragma unroll
      for (int r = 0; r < 8; ++r) {
        float v = c[r] + b0;
        if (col >= DZ) v = -softplus_f(v);
        alpha[(size_t)t * NB * DA + (size_t)(r0 + r) * DA + col] = v;
      }
    }
    __syncthreads();
    _Float16* tmp = hPrev; hPrev = hCur; hCur = tmp;
    p ^= 1;
  }
}

// --------------------------------------------------------------------------
// Kernel 3: forward scan, single persistent WG, async double-buffered alpha.
// --------------------------------------------------------------------------
__global__ __launch_bounds__(512, 1)
void forward_kernel(const float* __restrict__ u,          // (N,T,8)
                    const _Float16* __restrict__ w1T,     // [256][96]
                    const float* __restrict__ b1,
                    const _Float16* __restrict__ w2T,     // [64][256]
                    const float* __restrict__ b2,
                    const float* __restrict__ q_raw,
                    const float* __restrict__ m0,
                    const float* __restrict__ v0,
                    const float* __restrict__ alpha,      // (T,N,128) f32
                    float* __restrict__ out) {            // (N,T,256) f32
  extern __shared__ _Float16 smem[];
  _Float16* sW1 = smem;                  // 256*96
  _Float16* sW2 = sW1 + 24576;           // 64*256
  _Float16* sX  = sW2 + 16384;           // 32*96
  _Float16* sHt = sX + 3072;             // 32*256
  float* sM = (float*)(sHt + 8192);      // 32*64
  float* sV = sM + 2048;                 // 32*64
  float* sQ = sV + 2048;                 // 64
  float* sAl[2] = {sQ + 64, sQ + 64 + 4096};   // 2 x (32*128) f32
  const int tid = threadIdx.x, wave = tid >> 5, lane = tid & 31;

  // stage one (32 x 128) f32 alpha slice: thread owns 8 floats (2 x b128)
  auto stage_alpha = [&](float* dst, int t) {
    const float* g = alpha + (size_t)t * NB * DA + tid * 8;
    copy16_g2l(dst + tid * 8, g);
    copy16_g2l(dst + tid * 8 + 4, g + 4);
  };

  for (int i = tid; i < 24576; i += 512) sW1[i] = w1T[i];
  for (int i = tid; i < 16384; i += 512) sW2[i] = w2T[i];
  for (int i = tid; i < 2048; i += 512) { sM[i] = m0[i & 63]; sV[i] = v0[i & 63]; }
  if (tid < 64) sQ[tid] = softplus_f(q_raw[tid]);

  stage_alpha(sAl[0], 0);
  int p = 0;

  for (int t = 0; t < TT; ++t) {
    // build x = [m, u_t, 0pad] as f16 [32][96]
    for (int i = tid; i < NB * KDYN; i += 512) {
      int n = i / KDYN, k = i - n * KDYN;
      float v;
      if (k < DZ)           v = sM[n * DZ + k];
      else if (k < DZ + DU) v = u[(size_t)n * TT * DU + (size_t)t * DU + (k - DZ)];
      else                  v = 0.f;
      sX[i] = (_Float16)v;
    }
    if (t + 1 < TT) { stage_alpha(sAl[1 - p], t + 1); WAIT_ASYNC(2); }
    else            { WAIT_ASYNC(0); }
    __syncthreads();

    // GEMM1: 32x256, 32 tiles, 2 per wave, K=96
#pragma unroll
    for (int tt = 0; tt < 2; ++tt) {
      int ti = wave * 2 + tt;
      int rt = ti >> 4, ct = ti & 15;
      v8f c = {};
#pragma unroll
      for (int kk = 0; kk < 3; ++kk)
        c = wmma32(load_frag(sX + rt * 16 * KDYN + kk * 32, KDYN),
                   load_frag(sW1 + ct * 16 * KDYN + kk * 32, KDYN), c);
      int col = ct * 16 + (lane & 15);
      int r0  = rt * 16 + ((lane >> 4) << 3);
      float b0 = b1[col];
#pragma unroll
      for (int r = 0; r < 8; ++r)
        sHt[(r0 + r) * HDYN + col] = (_Float16)fast_tanh(c[r] + b0);
    }
    __syncthreads();

    // GEMM2 + Kalman-style update: 32x64 = 8 tiles, waves 0..7
    if (wave < 8) {
      int rt = wave >> 2, ct = wave & 3;
      v8f c = {};
#pragma unroll
      for (int kk = 0; kk < 8; ++kk)
        c = wmma32(load_frag(sHt + rt * 16 * HDYN + kk * 32, HDYN),
                   load_frag(sW2 + ct * 16 * HDYN + kk * 32, HDYN), c);
      int z  = ct * 16 + (lane & 15);
      int r0 = rt * 16 + ((lane >> 4) << 3);
      float b0 = b2[z], Qz = sQ[z];
#pragma unroll
      for (int r = 0; r < 8; ++r) {
        int n = r0 + r;
        float mp = c[r] + b0;
        float vp = sV[n * DZ + z] + Qz;
        float a1 = sAl[p][n * DA + z];
        float a2 = sAl[p][n * DA + DZ + z];
        float rvp = __builtin_amdgcn_rcpf(vp);
        float e1 = mp * rvp + a1;
        float e2 = a2 - 0.5f * rvp;
        float vs = -0.5f * __builtin_amdgcn_rcpf(e2);
        float ms = vs * e1;
        sM[n * DZ + z] = ms;
        sV[n * DZ + z] = vs;
        size_t base = (size_t)n * TT * 256 + (size_t)t * 256 + z;
        out[base]       = ms;
        out[base + 64]  = vs;
        out[base + 128] = mp;
        out[base + 192] = vp;
      }
    }
    __syncthreads();
    p ^= 1;
  }
}

// --------------------------------------------------------------------------
extern "C" void kernel_launch(void* const* d_in, const int* in_sizes, int n_in,
                              void* d_out, int out_size, void* d_ws, size_t ws_size,
                              hipStream_t stream) {
  const float* y      = (const float*)d_in[1];
  const float* u      = (const float*)d_in[2];
  const float* enc_w1 = (const float*)d_in[3];
  const float* enc_b1 = (const float*)d_in[4];
  const float* enc_w2 = (const float*)d_in[5];
  const float* enc_b2 = (const float*)d_in[6];
  const float* bw_wa  = (const float*)d_in[7];
  const float* bw_wh  = (const float*)d_in[8];
  const float* bw_b   = (const float*)d_in[9];
  const float* bw_wo  = (const float*)d_in[10];
  const float* bw_bo  = (const float*)d_in[11];
  const float* dyn_w1 = (const float*)d_in[12];
  const float* dyn_b1 = (const float*)d_in[13];
  const float* dyn_w2 = (const float*)d_in[14];
  const float* dyn_b2 = (const float*)d_in[15];
  const float* q_raw  = (const float*)d_in[16];
  const float* m0     = (const float*)d_in[17];
  const float* v0     = (const float*)d_in[18];
  float* out = (float*)d_out;

  char* ws = (char*)d_ws;
  _Float16*  bufA16 = (_Float16*)(ws);                   // 8 MB (N*T,128) f16
  float*     bufAl  = (float*)(ws + 16777216);           // 16 MB (T,N,128) f32
  _Float16*  w1T    = (_Float16*)(ws + 33554432);        // [256][128]
  _Float16*  w2T    = w1T  + 32768;                      // [128][256]
  _Float16*  bwaT   = w2T  + 32768;                      // [128][128]
  _Float16*  bwhT   = bwaT + 16384;
  _Float16*  bwoT   = bwhT + 16384;
  _Float16*  dw1T   = bwoT + 16384;                      // [256][96]
  _Float16*  dw2T   = dw1T + 24576;                      // [64][256]

  conv_transpose_f16<<<(256 * 128 + 255) / 256, 256, 0, stream>>>(enc_w1, w1T, 128, 256, 128);
  conv_transpose_f16<<<(128 * 256 + 255) / 256, 256, 0, stream>>>(enc_w2, w2T, 256, 128, 256);
  conv_transpose_f16<<<(128 * 128 + 255) / 256, 256, 0, stream>>>(bw_wa, bwaT, 128, 128, 128);
  conv_transpose_f16<<<(128 * 128 + 255) / 256, 256, 0, stream>>>(bw_wh, bwhT, 128, 128, 128);
  conv_transpose_f16<<<(128 * 128 + 255) / 256, 256, 0, stream>>>(bw_wo, bwoT, 128, 128, 128);
  conv_transpose_f16<<<(256 *  96 + 255) / 256, 256, 0, stream>>>(dyn_w1, dw1T, 72, 256, 96);
  conv_transpose_f16<<<( 64 * 256 + 255) / 256, 256, 0, stream>>>(dyn_w2, dw2T, 256, 64, 256);

  encoder_kernel<<<512, 512, 0, stream>>>(y, w1T, enc_b1, w2T, enc_b2, bufA16);

  size_t smem_bw = (size_t)(3 * 16384 + 2 * 4096 + 2 * 4096) * sizeof(_Float16); // 128 KB
  backward_kernel<<<1, 512, smem_bw, stream>>>(bufA16, bwaT, bwhT, bw_b, bwoT, bw_bo, bufAl);

  size_t smem_fw = (size_t)(24576 + 16384 + 3072 + 8192) * sizeof(_Float16)
                 + (size_t)(2048 + 2048 + 64 + 2 * 4096) * sizeof(float);        // ~150 KB
  forward_kernel<<<1, 512, smem_fw, stream>>>(u, dw1T, dyn_b1, dw2T, dyn_b2,
                                              q_raw, m0, v0, bufAl, out);
}
